// SMC_Transformer_16793322127637
// MI455X (gfx1250) — matrix operand order repaired and checked by reading, not compiled, not run on hardware
//
#include <hip/hip_runtime.h>
#include <hip/hip_bf16.h>
#include <math.h>

// ---------------------------------------------------------------------------
// CDNA5 (gfx1250) wave32 WMMA transformer block.
// Generic GEMM: C = epilogue( alpha * (A @ B + bias) ), bf16 WMMA, f32 accum.
// Block tile 128(M) x 64(N), 4 waves; each wave computes 32x64 with
// 2 A-frags x 4 B-frags = 8 WMMAs per 32-K step (12 ds_load_b128 / 8 WMMA).
// ---------------------------------------------------------------------------

typedef __attribute__((ext_vector_type(16))) __bf16 v16bf;
typedef __attribute__((ext_vector_type(8)))  __bf16 v8bf;
typedef __attribute__((ext_vector_type(4)))  __bf16 v4bf;
typedef __attribute__((ext_vector_type(8)))  float  v8f;
typedef __attribute__((ext_vector_type(4)))  float  v4f;

#define TILE_M 128
#define TILE_N 64
#define TILE_K 32
// row stride 40 elements = 80B: 16B aligned for b128 ops, bank step 20
// -> rows 0..15 map to distinct LDS banks (20*m mod 64 all distinct for m<16).
#define LDA_T  40   // A tile: [128 m][32 k] (k-contiguous per row)
#define LDB_T  40   // B tile: [ 64 n][32 k] (k-contiguous per row, i.e. B^T)

// Build fragments from LDS (2x ds_load_b128 each) and run 8 WMMAs.
__device__ __forceinline__ void frag_wmma(const __bf16* __restrict__ lA,
                                          const __bf16* __restrict__ lB,
                                          int wave, int lane, v8f acc[2][4])
{
  const int half = lane >> 4;   // 0: lanes 0-15, 1: lanes 16-31
  const int mrow = lane & 15;
  const int ncol = lane & 15;

  v16bf a[2];
  #pragma unroll
  for (int i2 = 0; i2 < 2; ++i2) {
    // A frag: lane elements are k = half*8..+7 and k = 16+half*8..+7
    const __bf16* pa = &lA[(wave * 32 + i2 * 16 + mrow) * LDA_T + half * 8];
    const v8bf a0 = *(const v8bf*)(pa);        // ds_load_b128
    const v8bf a1 = *(const v8bf*)(pa + 16);   // ds_load_b128
    a[i2] = __builtin_shufflevector(a0, a1, 0, 1, 2, 3, 4, 5, 6, 7,
                                    8, 9, 10, 11, 12, 13, 14, 15);
  }
  #pragma unroll
  for (int j = 0; j < 4; ++j) {
    // B frag: lane elements are k = half*8..+7 and k = 16+half*8..+7, fixed n
    const __bf16* pb = &lB[(j * 16 + ncol) * LDB_T + half * 8];
    const v8bf b0 = *(const v8bf*)(pb);        // ds_load_b128
    const v8bf b1 = *(const v8bf*)(pb + 16);   // ds_load_b128
    const v16bf bf = __builtin_shufflevector(b0, b1, 0, 1, 2, 3, 4, 5, 6, 7,
                                             8, 9, 10, 11, 12, 13, 14, 15);
    acc[0][j] = __builtin_amdgcn_wmma_f32_16x16x32_bf16(
        false, a[0], false, bf, (short)0, acc[0][j], false, false);
    acc[1][j] = __builtin_amdgcn_wmma_f32_16x16x32_bf16(
        false, a[1], false, bf, (short)0, acc[1][j], false, false);
  }
}

__launch_bounds__(128)
__global__ void gemm_bf16_wmma(const float* __restrict__ A,
                               const float* __restrict__ Bm,
                               const float* __restrict__ bias,
                               float* __restrict__ C,
                               int M, int N, int K,
                               long sA, long sB, long sC,
                               float alpha, int relu, int causal, int transB)
{
  alignas(16) __shared__ __bf16 lA[TILE_M * LDA_T];
  alignas(16) __shared__ __bf16 lB[TILE_N * LDB_T];

  const int tid  = threadIdx.x;
  const int lane = tid & 31;          // wave32
  const int wave = tid >> 5;          // 0..3

  const int bz = blockIdx.z;
  A  += (long)bz * sA;
  Bm += (long)bz * sB;
  C  += (long)bz * sC;

  const int tm = blockIdx.y * TILE_M;
  const int tn = blockIdx.x * TILE_N;

  v8f acc[2][4] = {};   // per wave: 32 rows x 64 cols of f32

  const int ksteps = (K + TILE_K - 1) / TILE_K;
  const bool fast = (tm + TILE_M <= M) && (tn + TILE_N <= N) &&
                    ((K & (TILE_K - 1)) == 0);

  if (fast) {
    // =================== branch-free fast loop ===================
    for (int ks = 0; ks < ksteps; ++ks) {
      const int k0 = ks * TILE_K;

      // stage A: thread = one row (tm+tid), 32 k: 8x v4f -> 4x ds_store_b128
      {
        const float* ga = &A[(long)(tm + tid) * K + k0];
        #pragma unroll
        for (int c = 0; c < 2; ++c) {
          const v4f f0 = *(const v4f*)(ga + c * 16 + 0);
          const v4f f1 = *(const v4f*)(ga + c * 16 + 4);
          const v4f f2 = *(const v4f*)(ga + c * 16 + 8);
          const v4f f3 = *(const v4f*)(ga + c * 16 + 12);
          v8bf h0, h1;
          #pragma unroll
          for (int i = 0; i < 4; ++i) {
            h0[i]     = (__bf16)f0[i];
            h0[4 + i] = (__bf16)f1[i];
            h1[i]     = (__bf16)f2[i];
            h1[4 + i] = (__bf16)f3[i];
          }
          *(v8bf*)&lA[tid * LDA_T + c * 16]     = h0;
          *(v8bf*)&lA[tid * LDA_T + c * 16 + 8] = h1;
        }
      }
      // stage B into n-major LDS
      if (!transB) {
        // B [K,N] row-major: 4x4 register-block transpose per thread.
        const int nb = (tid & 15) * 4;        // 0..60
        const int kb = (tid >> 4) * 4;        // 0..28
        const long base = (long)(k0 + kb) * N + tn + nb;
        const v4f r0 = *(const v4f*)&Bm[base + 0 * (long)N];
        const v4f r1 = *(const v4f*)&Bm[base + 1 * (long)N];
        const v4f r2 = *(const v4f*)&Bm[base + 2 * (long)N];
        const v4f r3 = *(const v4f*)&Bm[base + 3 * (long)N];
        #pragma unroll
        for (int j = 0; j < 4; ++j) {
          v4bf h;
          h[0] = (__bf16)r0[j];
          h[1] = (__bf16)r1[j];
          h[2] = (__bf16)r2[j];
          h[3] = (__bf16)r3[j];
          *(v4bf*)&lB[(nb + j) * LDB_T + kb] = h;   // ds_store_b64 along k
        }
      } else {
        // B [N,K] row-major (scores): matches n-major LDS directly.
        const int nn = tid >> 1;              // 0..63
        const int kc = (tid & 1) * 16;        // 0 or 16
        const float* gb = &Bm[(long)(tn + nn) * K + k0 + kc];
        const v4f f0 = *(const v4f*)(gb + 0);
        const v4f f1 = *(const v4f*)(gb + 4);
        const v4f f2 = *(const v4f*)(gb + 8);
        const v4f f3 = *(const v4f*)(gb + 12);
        v8bf h0, h1;
        #pragma unroll
        for (int i = 0; i < 4; ++i) {
          h0[i]     = (__bf16)f0[i];
          h0[4 + i] = (__bf16)f1[i];
          h1[i]     = (__bf16)f2[i];
          h1[4 + i] = (__bf16)f3[i];
        }
        *(v8bf*)&lB[nn * LDB_T + kc]     = h0;
        *(v8bf*)&lB[nn * LDB_T + kc + 8] = h1;
      }
      __syncthreads();

      // L2 prefetch of next A tile (global_prefetch_b8)
      if (ks + 1 < ksteps)
        __builtin_prefetch(&A[(long)(tm + tid) * K + k0 + TILE_K], 0, 1);

      frag_wmma(lA, lB, wave, lane, acc);
      __syncthreads();
    }
  } else {
    // =================== predicated edge loop ===================
    for (int ks = 0; ks < ksteps; ++ks) {
      const int k0 = ks * TILE_K;
      // stage A elementwise: thread = row
      {
        const int gr = tm + tid;
        #pragma unroll 4
        for (int u = 0; u < TILE_K; ++u) {
          const int gk = k0 + u;
          float fv = 0.0f;
          if (gr < M && gk < K) fv = A[(long)gr * K + gk];
          lA[tid * LDA_T + u] = (__bf16)fv;
        }
      }
      // stage B elementwise
      if (!transB) {
        const int nb = (tid & 15) * 4;
        const int kb = (tid >> 4) * 4;
        #pragma unroll
        for (int uk = 0; uk < 4; ++uk) {
          #pragma unroll
          for (int j = 0; j < 4; ++j) {
            const int gk = k0 + kb + uk;
            const int gn = tn + nb + j;
            float fv = 0.0f;
            if (gk < K && gn < N) fv = Bm[(long)gk * N + gn];
            lB[(nb + j) * LDB_T + kb + uk] = (__bf16)fv;
          }
        }
      } else {
        const int nn = tid >> 1;
        const int kc = (tid & 1) * 16;
        const int gn = tn + nn;
        #pragma unroll 4
        for (int u = 0; u < 16; ++u) {
          const int kk = kc + u;
          const int gk = k0 + kk;
          float fv = 0.0f;
          if (gn < N && gk < K) fv = Bm[(long)gn * K + gk];
          lB[nn * LDB_T + kk] = (__bf16)fv;
        }
      }
      __syncthreads();
      frag_wmma(lA, lB, wave, lane, acc);
      __syncthreads();
    }
  }

  // ---- epilogue: bias, alpha scale, causal mask, relu ----
  {
    const int half = lane >> 4;
    const int ncol = lane & 15;
    #pragma unroll
    for (int i2 = 0; i2 < 2; ++i2) {
      #pragma unroll
      for (int j = 0; j < 4; ++j) {
        #pragma unroll
        for (int r = 0; r < 8; ++r) {
          const int grow = tm + wave * 32 + i2 * 16 + r + half * 8; // C/D layout
          const int gcol = tn + j * 16 + ncol;
          if (grow < M && gcol < N) {
            float vv = acc[i2][j][r];
            if (bias) vv += bias[gcol];
            vv *= alpha;
            if (causal && gcol > grow) vv = -1e9f;
            if (relu) vv = fmaxf(vv, 0.0f);
            C[(long)grow * N + gcol] = vv;
          }
        }
      }
    }
  }
}

// ---------------------------------------------------------------------------
// Row softmax (in place), one 256-thread block per row.
// ---------------------------------------------------------------------------
__launch_bounds__(256)
__global__ void softmax_rows(float* __restrict__ attn, int cols)
{
  __shared__ float red[256];
  float* p = attn + (long)blockIdx.x * cols;

  float mx = -3.0e38f;
  for (int c = threadIdx.x; c < cols; c += 256) mx = fmaxf(mx, p[c]);
  red[threadIdx.x] = mx; __syncthreads();
  for (int s = 128; s > 0; s >>= 1) {
    if (threadIdx.x < s) red[threadIdx.x] = fmaxf(red[threadIdx.x], red[threadIdx.x + s]);
    __syncthreads();
  }
  mx = red[0]; __syncthreads();

  float sum = 0.f;
  for (int c = threadIdx.x; c < cols; c += 256) {
    float e = __expf(p[c] - mx);
    p[c] = e;
    sum += e;
  }
  red[threadIdx.x] = sum; __syncthreads();
  for (int s = 128; s > 0; s >>= 1) {
    if (threadIdx.x < s) red[threadIdx.x] += red[threadIdx.x + s];
    __syncthreads();
  }
  const float inv = 1.0f / red[0];
  for (int c = threadIdx.x; c < cols; c += 256) p[c] *= inv;
}

// ---------------------------------------------------------------------------
// out = LayerNorm(a + b) * g + beta   (population variance, eps 1e-6)
// ---------------------------------------------------------------------------
__launch_bounds__(256)
__global__ void add_layernorm(const float* __restrict__ a, const float* __restrict__ b,
                              const float* __restrict__ g, const float* __restrict__ beta,
                              float* __restrict__ out, int D)
{
  __shared__ float rs[256], rs2[256];
  const long row = blockIdx.x;
  const float* pa = a + row * (long)D;
  const float* pb = b + row * (long)D;

  float s = 0.f, s2 = 0.f;
  for (int c = threadIdx.x; c < D; c += 256) {
    const float x = pa[c] + pb[c];
    s += x; s2 += x * x;
  }
  rs[threadIdx.x] = s; rs2[threadIdx.x] = s2; __syncthreads();
  for (int st = 128; st > 0; st >>= 1) {
    if (threadIdx.x < st) { rs[threadIdx.x] += rs[threadIdx.x + st]; rs2[threadIdx.x] += rs2[threadIdx.x + st]; }
    __syncthreads();
  }
  const float mean = rs[0] / (float)D;
  const float var  = rs2[0] / (float)D - mean * mean;
  const float inv  = rsqrtf(var + 1e-6f);
  for (int c = threadIdx.x; c < D; c += 256) {
    const float x = pa[c] + pb[c];
    out[row * (long)D + c] = g[c] * (x - mean) * inv + beta[c];
  }
}

// ---------------------------------------------------------------------------
extern "C" void kernel_launch(void* const* d_in, const int* in_sizes, int n_in,
                              void* d_out, int out_size, void* d_ws, size_t ws_size,
                              hipStream_t stream) {
  (void)in_sizes; (void)n_in; (void)out_size; (void)ws_size;

  constexpr int Bb = 8, S = 2048, FX = 16, FY = 16, D = 512, DFF = 2048;
  constexpr long BS = (long)Bb * S;               // 16384 rows

  const float* inputs = (const float*)d_in[0];
  // d_in[1] = targets (unused)
  const float* Wp  = (const float*)d_in[2];   const float* bp   = (const float*)d_in[3];
  const float* Wq  = (const float*)d_in[4];   const float* bq   = (const float*)d_in[5];
  const float* Wk  = (const float*)d_in[6];   const float* bk   = (const float*)d_in[7];
  const float* Wv  = (const float*)d_in[8];   const float* bv   = (const float*)d_in[9];
  const float* Wo  = (const float*)d_in[10];  const float* bo   = (const float*)d_in[11];
  const float* g1  = (const float*)d_in[12];  const float* be1  = (const float*)d_in[13];
  const float* g2  = (const float*)d_in[14];  const float* be2  = (const float*)d_in[15];
  const float* Wf1 = (const float*)d_in[16];  const float* bf1  = (const float*)d_in[17];
  const float* Wf2 = (const float*)d_in[18];  const float* bf2  = (const float*)d_in[19];
  const float* Wfin= (const float*)d_in[20];  const float* bfin = (const float*)d_in[21];

  // d_out layout: pred | pred2 | k | v | r | attn
  float* out   = (float*)d_out;
  float* pred  = out;
  float* pred2 = pred + BS * FY;
  float* kout  = pred2 + BS * FY;
  float* vout  = kout + BS * D;
  float* rout  = vout + BS * D;
  float* attn  = rout + BS * D;

  // workspace: x | q(->zo->ff) | z | out1 | h
  float* ws = (float*)d_ws;
  float* x  = ws;
  float* q  = x  + BS * D;
  float* z  = q  + BS * D;
  float* o1 = z  + BS * D;
  float* h  = o1 + BS * D;

  const dim3 blk(128, 1, 1);
  auto grid = [](long M, long N, int Z) {
    return dim3((unsigned)((N + TILE_N - 1) / TILE_N),
                (unsigned)((M + TILE_M - 1) / TILE_M), (unsigned)Z);
  };

  const float sqrtd    = sqrtf((float)D);
  const float invsqrtd = 1.0f / sqrtd;

  // 1) x = (inputs @ Wp + bp) * sqrt(D)        [16384,16] x [16,512]
  gemm_bf16_wmma<<<grid(BS, D, 1), blk, 0, stream>>>(
      inputs, Wp, bp, x, (int)BS, D, FX, 0, 0, 0, sqrtd, 0, 0, 0);

  // 2) q / k / v = x @ W{q,k,v} + b            [16384,512] x [512,512]
  gemm_bf16_wmma<<<grid(BS, D, 1), blk, 0, stream>>>(
      x, Wq, bq, q, (int)BS, D, D, 0, 0, 0, 1.0f, 0, 0, 0);
  gemm_bf16_wmma<<<grid(BS, D, 1), blk, 0, stream>>>(
      x, Wk, bk, kout, (int)BS, D, D, 0, 0, 0, 1.0f, 0, 0, 0);
  gemm_bf16_wmma<<<grid(BS, D, 1), blk, 0, stream>>>(
      x, Wv, bv, vout, (int)BS, D, D, 0, 0, 0, 1.0f, 0, 0, 0);

  // 3) scores = causal_mask(q @ k^T / sqrt(D)) per batch -> attn buffer
  gemm_bf16_wmma<<<grid(S, S, Bb), blk, 0, stream>>>(
      q, kout, nullptr, attn, S, S, D,
      (long)S * D, (long)S * D, (long)S * S, invsqrtd, 0, /*causal=*/1, /*transB=*/1);

  // 4) softmax over each of the 16384 rows of length 2048 (in place)
  softmax_rows<<<dim3((unsigned)BS, 1, 1), dim3(256, 1, 1), 0, stream>>>(attn, S);

  // 5) z = attn @ v per batch                  [2048,2048] x [2048,512]
  gemm_bf16_wmma<<<grid(S, D, Bb), blk, 0, stream>>>(
      attn, vout, nullptr, z, S, D, S,
      (long)S * S, (long)S * D, (long)S * D, 1.0f, 0, 0, 0);

  // 6) zo = z @ Wo + bo   (reuse q buffer)
  gemm_bf16_wmma<<<grid(BS, D, 1), blk, 0, stream>>>(
      z, Wo, bo, q, (int)BS, D, D, 0, 0, 0, 1.0f, 0, 0, 0);

  // 7) out1 = LN(zo + x)
  add_layernorm<<<dim3((unsigned)BS, 1, 1), dim3(256, 1, 1), 0, stream>>>(
      q, x, g1, be1, o1, D);

  // 8) h = relu(out1 @ Wf1 + bf1)              [16384,512] x [512,2048]
  gemm_bf16_wmma<<<grid(BS, DFF, 1), blk, 0, stream>>>(
      o1, Wf1, bf1, h, (int)BS, DFF, D, 0, 0, 0, 1.0f, /*relu=*/1, 0, 0);

  // 9) ff = h @ Wf2 + bf2  (reuse q buffer)    [16384,2048] x [2048,512]
  gemm_bf16_wmma<<<grid(BS, D, 1), blk, 0, stream>>>(
      h, Wf2, bf2, q, (int)BS, D, DFF, 0, 0, 0, 1.0f, 0, 0, 0);

  // 10) r = LN(ff + out1) -> d_out r region
  add_layernorm<<<dim3((unsigned)BS, 1, 1), dim3(256, 1, 1), 0, stream>>>(
      q, o1, g2, be2, rout, D);

  // 11) pred = r @ Wfin + bfin                 [16384,512] x [512,16]
  gemm_bf16_wmma<<<grid(BS, FY, 1), blk, 0, stream>>>(
      rout, Wfin, bfin, pred, (int)BS, FY, D, 0, 0, 0, 1.0f, 0, 0, 0);

  // 12) pred_resampl == pred (no resampling path)
  hipMemcpyAsync(pred2, pred, (size_t)(BS * FY) * sizeof(float),
                 hipMemcpyDeviceToDevice, stream);
}
